// RNNDecoder_31061203485197
// MI455X (gfx1250) — matrix operand order repaired and checked by reading, not compiled, not run on hardware
//
#include <hip/hip_runtime.h>

typedef __attribute__((ext_vector_type(16))) _Float16 v16h;
typedef __attribute__((ext_vector_type(8)))  _Float16 h8v;
typedef __attribute__((ext_vector_type(8)))  float    v8f;

#define WMMA_F16(a,b,c) __builtin_amdgcn_wmma_f32_16x16x32_f16(false,(a),false,(b),(short)0,(c),false,false)

__device__ __forceinline__ v16h cat8(h8v lo, h8v hi) {
  return __builtin_shufflevector(lo, hi, 0,1,2,3,4,5,6,7,8,9,10,11,12,13,14,15);
}
__device__ __forceinline__ float sigmoidf_(float x) { return 1.f / (1.f + expf(-x)); }

// ---------------------------------------------------------------------------
// Prep: f32 -> f16 copy (LSTM weights, layout preserved [1024,256])
// ---------------------------------------------------------------------------
__global__ void cvt_f16_kernel(const float* __restrict__ s, _Float16* __restrict__ d, int n) {
  int i = blockIdx.x * 256 + threadIdx.x;
  if (i < n) d[i] = (_Float16)s[i];
}

// ---------------------------------------------------------------------------
// Prep: pack conv weight [Cout][Cin][KH][KW] f32 -> [CoutPad][Cin*KH*KW] f16,
// kernel flip (for convT) baked in; rows co>=Cout are zero.
// ---------------------------------------------------------------------------
__global__ void pack_wgt_kernel(const float* __restrict__ w, _Float16* __restrict__ d,
                                int Cout, int CoutPad, int Ktot, int Cin, int KH, int KW, int flip) {
  int i = blockIdx.x * 256 + threadIdx.x;
  if (i >= CoutPad * Ktot) return;
  int co = i / Ktot, k = i - co * Ktot;
  float v = 0.f;
  if (co < Cout) {
    int KHW = KH * KW;
    int ci = k / KHW; int rr = k - ci * KHW; int kh = rr / KW; int kw = rr - kh * KW;
    if (flip) { kh = KH - 1 - kh; kw = KW - 1 - kw; }
    v = w[(((long)co * Cin + ci) * KH + kh) * KW + kw];
  }
  d[i] = (_Float16)v;
}

// ---------------------------------------------------------------------------
// Stage 0: style MLP  ss = relu(style@w1^T+b1)@w2^T+b2
// ---------------------------------------------------------------------------
__global__ void mlp_kernel(const float* __restrict__ style,
                           const float* __restrict__ w1, const float* __restrict__ b1,
                           const float* __restrict__ w2, const float* __restrict__ b2,
                           float* __restrict__ ss) {
  __shared__ float hid[8 * 144];
  for (int i = threadIdx.x; i < 8 * 144; i += 256) {
    int b = i / 144, j = i % 144;
    float a = b1[j];
    for (int k = 0; k < 256; ++k) a += style[b * 256 + k] * w1[j * 256 + k];
    hid[i] = fmaxf(a, 0.f);
  }
  __syncthreads();
  for (int i = threadIdx.x; i < 8 * 32; i += 256) {
    int b = i / 32, j = i % 32;
    float a = b2[j];
    for (int k = 0; k < 144; ++k) a += hid[b * 144 + k] * w2[j * 144 + k];
    ss[i] = a;
  }
}

// ---------------------------------------------------------------------------
// Stage 1: concat(content, broadcast ss) -> 1x1 conv (112->256) -> relu
// ---------------------------------------------------------------------------
__global__ void embed_kernel(const float* __restrict__ content, const float* __restrict__ ss,
                             const float* __restrict__ ew, const float* __restrict__ eb,
                             float* __restrict__ inp) {
  int idx = blockIdx.x * 256 + threadIdx.x;
  if (idx >= 8 * 256 * 16) return;
  int l = idx % 16;
  int e = (idx / 16) % 256;
  int b = idx / (16 * 256);
  float a = eb[e];
  for (int i = 0; i < 80; ++i) a += ew[e * 112 + i] * content[(b * 80 + i) * 16 + l];
  for (int i = 0; i < 32; ++i) a += ew[e * 112 + 80 + i] * ss[b * 32 + i];
  inp[idx] = fmaxf(a, 0.f);
}

// ---------------------------------------------------------------------------
// WMMA LSTM gate GEMM (512 threads = 16 waves, 4 N-tiles per wave):
//   g[8][1024] = x[8,256]@wih^T + h[8,256]@whh^T + bih + bhh
// x/h live in LDS as f16 [16][256] (rows 8..15 zero). Weights pre-converted f16.
// Rolled K-loop (2 WMMAs + 8 b128 halves per iter) to avoid VGPR spills.
// ---------------------------------------------------------------------------
__device__ void lstm_gates_wmma(const _Float16* s_x, const _Float16* s_h,
                                const _Float16* __restrict__ wih, const _Float16* __restrict__ whh,
                                const float* __restrict__ bih, const float* __restrict__ bhh,
                                float* s_g, int tid) {
  int lane = tid & 31, wave = tid >> 5;          // 16 waves
  int half = lane >> 4, l15 = lane & 15;
  const _Float16* xrow = s_x + l15 * 256;
  const _Float16* hrow = s_h + l15 * 256;
  for (int t = 0; t < 4; ++t) {
    int n0 = (wave * 4 + t) * 16;
    const _Float16* wi = wih + (long)(n0 + l15) * 256;
    const _Float16* wh = whh + (long)(n0 + l15) * 256;
    v8f acc = {};
#pragma unroll 1
    for (int kk = 0; kk < 256; kk += 32) {
      v16h a = cat8(*(const h8v*)(xrow + kk + half * 8),
                    *(const h8v*)(xrow + kk + 16 + half * 8));
      v16h b = cat8(*(const h8v*)(wi + kk + half * 8),
                    *(const h8v*)(wi + kk + 16 + half * 8));
      acc = WMMA_F16(a, b, acc);
      v16h a2 = cat8(*(const h8v*)(hrow + kk + half * 8),
                     *(const h8v*)(hrow + kk + 16 + half * 8));
      v16h b2 = cat8(*(const h8v*)(wh + kk + half * 8),
                     *(const h8v*)(wh + kk + 16 + half * 8));
      acc = WMMA_F16(a2, b2, acc);
    }
    if (half == 0) {   // rows 0..7 = real batch rows; wave-half-uniform branch
      int n = n0 + l15;
      float bs = bih[n] + bhh[n];
#pragma unroll
      for (int r = 0; r < 8; ++r) s_g[r * 1024 + n] = acc[r] + bs;
    }
  }
}

// ---------------------------------------------------------------------------
// Stage 2: spacing_rnn — 192 sequential attention-shift LSTM steps, one block.
// ---------------------------------------------------------------------------
__global__ __launch_bounds__(512) void spacing_rnn_kernel(
    const float* __restrict__ inp,          // [8,256,16]
    const _Float16* __restrict__ wih, const _Float16* __restrict__ whh,
    const float* __restrict__ bih, const float* __restrict__ bhh,
    const float* __restrict__ shift_w, const float* __restrict__ shift_b,
    float* __restrict__ hs)                 // [192,8,256]
{
  __shared__ alignas(16) _Float16 s_tin[16 * 256];
  __shared__ alignas(16) _Float16 s_h[16 * 256];
  __shared__ float s_c[8 * 256];
  __shared__ float s_g[8 * 1024];
  __shared__ float s_att[8 * 16];
  __shared__ float s_sp[8 * 2];
  int tid = threadIdx.x;

  for (int i = tid; i < 16 * 256; i += 512) { s_tin[i] = (_Float16)0.f; s_h[i] = (_Float16)0.f; }
  for (int i = tid; i < 8 * 256; i += 512) s_c[i] = 0.f;
  for (int i = tid; i < 8 * 16; i += 512) s_att[i] = ((i % 16) == 0) ? 1.f : 0.f;
  __syncthreads();

  for (int step = 0; step < 192; ++step) {
    for (int i = tid; i < 8 * 256; i += 512) {
      int b = i >> 8, e = i & 255;
      float acc = 0.f;
      for (int l = 0; l < 16; ++l) acc += inp[(b * 256 + e) * 16 + l] * s_att[b * 16 + l];
      s_tin[i] = (_Float16)acc;
    }
    __syncthreads();

    lstm_gates_wmma(s_tin, s_h, wih, whh, bih, bhh, s_g, tid);
    __syncthreads();

    for (int i = tid; i < 8 * 256; i += 512) {
      int b = i >> 8, e = i & 255;
      float gi = s_g[b * 1024 + e];
      float gf = s_g[b * 1024 + 256 + e];
      float gg = s_g[b * 1024 + 512 + e];
      float go = s_g[b * 1024 + 768 + e];
      float c = sigmoidf_(gf) * s_c[i] + sigmoidf_(gi) * tanhf(gg);
      float h = sigmoidf_(go) * tanhf(c);
      s_c[i] = c;
      s_h[i] = (_Float16)h;
      hs[(long)step * 2048 + i] = h;
    }
    __syncthreads();

    if (tid < 16) {
      int b = tid >> 1, j = tid & 1;
      float acc = shift_b[j];
      for (int e = 0; e < 256; ++e) acc += (float)s_h[b * 256 + e] * shift_w[j * 256 + e];
      s_sp[b * 2 + j] = acc;
    }
    __syncthreads();

    if (tid < 8) {
      int b = tid;
      float s = sigmoidf_(s_sp[b * 2 + 0]);
      float sharp = fmaxf(s_sp[b * 2 + 1], 0.f) + 1.f;
      float prev = 0.f, tmp[16], sum = 0.f;
      for (int l = 0; l < 16; ++l) {
        float a = s_att[b * 16 + l];
        float na = (1.f - s) * a + s * prev;
        prev = a;
        tmp[l] = powf(na, sharp);
        sum += tmp[l];
      }
      float inv = 1.f / sum;
      for (int l = 0; l < 16; ++l) s_att[b * 16 + l] = tmp[l] * inv;
    }
    __syncthreads();
  }
}

// ---------------------------------------------------------------------------
// Stage 3: bidirectional LSTM. block 0 = forward, block 1 = backward.
// ---------------------------------------------------------------------------
__global__ __launch_bounds__(512) void bilstm_kernel(
    const float* __restrict__ hs,
    const _Float16* __restrict__ wihf, const _Float16* __restrict__ whhf,
    const float* __restrict__ bihf, const float* __restrict__ bhhf,
    const _Float16* __restrict__ wihb, const _Float16* __restrict__ whhb,
    const float* __restrict__ bihb, const float* __restrict__ bhhb,
    float* __restrict__ hf, float* __restrict__ hb)
{
  __shared__ alignas(16) _Float16 s_x[16 * 256];
  __shared__ alignas(16) _Float16 s_h[16 * 256];
  __shared__ float s_c[8 * 256];
  __shared__ float s_g[8 * 1024];
  int tid = threadIdx.x;
  int dir = blockIdx.x;
  const _Float16* wih = dir ? wihb : wihf;
  const _Float16* whh = dir ? whhb : whhf;
  const float* bih = dir ? bihb : bihf;
  const float* bhh = dir ? bhhb : bhhf;
  float* hout = dir ? hb : hf;

  for (int i = tid; i < 16 * 256; i += 512) { s_x[i] = (_Float16)0.f; s_h[i] = (_Float16)0.f; }
  for (int i = tid; i < 8 * 256; i += 512) s_c[i] = 0.f;
  __syncthreads();

  for (int t = 0; t < 192; ++t) {
    int src = dir ? (191 - t) : t;
    for (int i = tid; i < 8 * 256; i += 512) s_x[i] = (_Float16)hs[(long)src * 2048 + i];
    __syncthreads();
    lstm_gates_wmma(s_x, s_h, wih, whh, bih, bhh, s_g, tid);
    __syncthreads();
    for (int i = tid; i < 8 * 256; i += 512) {
      int b = i >> 8, e = i & 255;
      float gi = s_g[b * 1024 + e];
      float gf = s_g[b * 1024 + 256 + e];
      float gg = s_g[b * 1024 + 512 + e];
      float go = s_g[b * 1024 + 768 + e];
      float c = sigmoidf_(gf) * s_c[i] + sigmoidf_(gi) * tanhf(gg);
      float h = sigmoidf_(go) * tanhf(c);
      s_c[i] = c;
      s_h[i] = (_Float16)h;
      hout[(long)src * 2048 + i] = h;
    }
    __syncthreads();
  }
}

// ---------------------------------------------------------------------------
// Stage 4: pack d = transpose(concat(hf,hb)) -> [8,256,1,384]
// ---------------------------------------------------------------------------
__global__ void build_d_kernel(const float* __restrict__ hf, const float* __restrict__ hb,
                               float* __restrict__ dten) {
  int idx = blockIdx.x * 256 + threadIdx.x;
  if (idx >= 8 * 256 * 384) return;
  int w = idx % 384;
  int c = (idx / 384) % 256;
  int b = idx / (384 * 256);
  int e2 = 2 * c + (w >= 192 ? 1 : 0);
  int l = w % 192;
  float v = (e2 < 256) ? hf[((long)l * 8 + b) * 256 + e2]
                       : hb[((long)l * 8 + b) * 256 + (e2 - 256)];
  dten[idx] = v;
}

// ---------------------------------------------------------------------------
// Block-cooperative implicit-GEMM conv via WMMA.
// grid = (Mtiles, ceil(NB/8)); block = 256 threads (8 waves).
// A 16x32 f16 im2col slice is staged in LDS per K-step (branchless clamped
// gather, pixel decomposition hoisted out of the K-loop); each wave owns one
// 16-channel N-tile and streams its packed f16 weight row with b128 loads.
// flags: bit1 = input nearest-2x upsample, bits2-3 = act (0 none,1 relu,2 tanh).
// dilH/dilW = lhs_dilation (convT-as-conv). Kernel flip already baked in wpk.
// ---------------------------------------------------------------------------
__global__ __launch_bounds__(256) void conv_wmma_kernel(
    const float* __restrict__ in, const _Float16* __restrict__ wpk,
    const float* __restrict__ bias, const float* __restrict__ resid,
    float* __restrict__ out,
    int B, int Cin, int Hin, int Win, int Cout, int Hout, int Wout,
    int KH, int KW, int padH, int padW, int dilH, int dilW, int flags, int NB)
{
  __shared__ alignas(16) _Float16 sA[16 * 32];
  int mtile = blockIdx.x;
  int wv = threadIdx.x >> 5;
  int lane = threadIdx.x & 31;
  int half = lane >> 4, l15 = lane & 15;
  int ntile = blockIdx.y * 8 + wv;
  int ntileEff = ntile < NB ? ntile : NB - 1;
  int KHW = KH * KW;
  int Ktot = Cin * KHW;
  long HWo = (long)Hout * Wout;
  bool ups = flags & 2;
  int act = (flags >> 2) & 3;

  // staging coordinates: thread covers rows (mA, mA+8) at column kloc
  int kloc = threadIdx.x & 31;
  int mA = threadIdx.x >> 5;                  // 0..7
  int bAx[2], ohx[2], owx[2];
#pragma unroll
  for (int rep = 0; rep < 2; ++rep) {
    long pA = (long)mtile * 16 + mA + rep * 8;
    int bA = (int)(pA / HWo);
    int pr = (int)(pA - (long)bA * HWo);
    bAx[rep] = bA;
    ohx[rep] = pr / Wout;
    owx[rep] = pr - ohx[rep] * Wout;
  }
  const _Float16* wrow = wpk + (long)(ntileEff * 16 + l15) * Ktot;

  v8f acc = {};
#pragma unroll 1
  for (int kk = 0; kk < Ktot; kk += 32) {
    // ---- cooperative A staging: 16 rows x 32 K of f16 ----
    int k = kk + kloc;
    int ci = k / KHW; int rr = k - ci * KHW; int kh = rr / KW; int kw = rr - kh * KW;
#pragma unroll
    for (int rep = 0; rep < 2; ++rep) {
      int ihd = ohx[rep] + kh - padH, iwd = owx[rep] + kw - padW;
      int ih, iw; bool ok;
      if (ups) {
        ok = (ihd >= 0) & (ihd < 2 * Hin) & (iwd >= 0) & (iwd < 2 * Win);
        ih = ihd >> 1; iw = iwd >> 1;
      } else {
        ok = (ihd >= 0) & (iwd >= 0) & (ihd % dilH == 0) & (iwd % dilW == 0);
        ih = ihd / dilH; iw = iwd / dilW;
        ok = ok & (ih < Hin) & (iw < Win);
      }
      int ihc = ih < 0 ? 0 : (ih >= Hin ? Hin - 1 : ih);
      int iwc = iw < 0 ? 0 : (iw >= Win ? Win - 1 : iw);
      float v = in[(((long)bAx[rep] * Cin + ci) * Hin + ihc) * Win + iwc];  // always in-bounds
      sA[(mA + rep * 8) * 32 + kloc] = (_Float16)(ok ? v : 0.f);
    }
    __syncthreads();
    // ---- fragments: 2x ds_load_b128 (A) + 2x global b128 (B) ----
    if (kk + 128 < Ktot) __builtin_prefetch(wrow + kk + 128, 0, 1);
    v16h a = cat8(*(const h8v*)(sA + l15 * 32 + half * 8),
                  *(const h8v*)(sA + l15 * 32 + 16 + half * 8));
    v16h b = cat8(*(const h8v*)(wrow + kk + half * 8),
                  *(const h8v*)(wrow + kk + 16 + half * 8));
    acc = WMMA_F16(a, b, acc);
    __syncthreads();
  }

  if (ntile < NB) {
    int n = ntile * 16 + l15;
    if (n < Cout) {
      float bs = bias[n];
#pragma unroll
      for (int r = 0; r < 8; ++r) {
        long p = (long)mtile * 16 + r + 8 * half;
        int b2 = (int)(p / HWo); int pr = (int)(p - (long)b2 * HWo);
        int oh = pr / Wout; int ow = pr - oh * Wout;
        long oidx = (((long)b2 * Cout + n) * Hout + oh) * Wout + ow;
        float v = acc[r] + bs;
        if (resid) v += resid[oidx];
        if (act == 1) v = fmaxf(v, 0.f);
        else if (act == 2) v = tanhf(v);
        out[oidx] = v;
      }
    }
  }
}

// ---------------------------------------------------------------------------
// Per-sample LayerNorm over (C,H,W), unbiased std; apply fused with relu.
// ---------------------------------------------------------------------------
__global__ void ln_stats_kernel(const float* __restrict__ x, float* __restrict__ stat, int n) {
  __shared__ float ssum[256], ssq[256];
  const float* xb = x + (long)blockIdx.x * n;
  float s = 0.f, q = 0.f;
  for (int i = threadIdx.x; i < n; i += 256) { float v = xb[i]; s += v; q += v * v; }
  ssum[threadIdx.x] = s; ssq[threadIdx.x] = q;
  __syncthreads();
  for (int st = 128; st > 0; st >>= 1) {
    if (threadIdx.x < st) { ssum[threadIdx.x] += ssum[threadIdx.x + st]; ssq[threadIdx.x] += ssq[threadIdx.x + st]; }
    __syncthreads();
  }
  if (threadIdx.x == 0) {
    float m = ssum[0] / n;
    float var = (ssq[0] - (float)n * m * m) / (float)(n - 1);
    stat[blockIdx.x * 2] = m;
    stat[blockIdx.x * 2 + 1] = sqrtf(fmaxf(var, 0.f));
  }
}

__global__ void ln_apply_kernel(float* __restrict__ x, const float* __restrict__ stat,
                                const float* __restrict__ g, const float* __restrict__ bta,
                                int HW, int n) {
  long idx = (long)blockIdx.x * 256 + threadIdx.x;
  long tot = (long)8 * n;
  if (idx >= tot) return;
  int b = (int)(idx / n);
  int r = (int)(idx % n);
  int c = r / HW;
  float m = stat[b * 2], sd = stat[b * 2 + 1];
  float v = (x[idx] - m) / (sd + 1e-5f) * g[c] + bta[c];
  x[idx] = fmaxf(v, 0.f);
}

// ---------------------------------------------------------------------------
// Host-side orchestration
// ---------------------------------------------------------------------------
static void launch_conv(hipStream_t st, const float* in, const _Float16* wpk, const float* bias,
                        const float* resid, float* out,
                        int B, int Cin, int Hin, int Win, int Cout, int Hout, int Wout,
                        int KH, int KW, int padH, int padW, int dilH, int dilW,
                        int ups, int act) {
  long Mtot = (long)B * Hout * Wout;
  int Mtiles = (int)((Mtot + 15) / 16);
  int NB = (Cout + 15) / 16;
  int gy = (NB + 7) / 8;
  int flags = (ups ? 2 : 0) | (act << 2);
  conv_wmma_kernel<<<dim3(Mtiles, gy), 256, 0, st>>>(in, wpk, bias, resid, out,
                                                     B, Cin, Hin, Win, Cout, Hout, Wout,
                                                     KH, KW, padH, padW, dilH, dilW, flags, NB);
}

static void launch_pack(hipStream_t st, const float* w, _Float16* d,
                        int Cout, int Cin, int KH, int KW, int flip) {
  int CoutPad = ((Cout + 15) / 16) * 16;
  int Ktot = Cin * KH * KW;
  int tot = CoutPad * Ktot;
  pack_wgt_kernel<<<(tot + 255) / 256, 256, 0, st>>>(w, d, Cout, CoutPad, Ktot, Cin, KH, KW, flip);
}

extern "C" void kernel_launch(void* const* d_in, const int* in_sizes, int n_in,
                              void* d_out, int out_size, void* d_ws, size_t ws_size,
                              hipStream_t stream) {
  (void)in_sizes; (void)n_in; (void)out_size; (void)ws_size;
#define F(i) ((const float*)d_in[(i)])
  const float* content = F(0);
  const float* style   = F(1);
  int p = 2;
  const float* sp_w1 = F(p++); const float* sp_b1 = F(p++);
  const float* sp_w2 = F(p++); const float* sp_b2 = F(p++);
  const float* emb_w = F(p++); const float* emb_b = F(p++);
  const float* r1_wih = F(p++); const float* r1_whh = F(p++);
  const float* r1_bih = F(p++); const float* r1_bhh = F(p++);
  const float* shift_w = F(p++); const float* shift_b = F(p++);
  const float* r2f_wih = F(p++); const float* r2f_whh = F(p++);
  const float* r2f_bih = F(p++); const float* r2f_bhh = F(p++);
  const float* r2b_wih = F(p++); const float* r2b_whh = F(p++);
  const float* r2b_bih = F(p++); const float* r2b_bhh = F(p++);
  const float* c0_w = F(p++); const float* c0_b = F(p++);
  const float* c1_w = F(p++); const float* c1_b = F(p++);
  const float* c2_w = F(p++); const float* c2_b = F(p++);
  const float* res_w1[6]; const float* res_b1[6]; const float* res_w2[6]; const float* res_b2[6];
  for (int s = 0; s < 6; ++s) { res_w1[s] = F(p++); res_b1[s] = F(p++); res_w2[s] = F(p++); res_b2[s] = F(p++); }
  const float* c3_w = F(p++); const float* c3_b = F(p++);
  const float* ln3_g = F(p++); const float* ln3_b = F(p++);
  const float* c4_w = F(p++); const float* c4_b = F(p++);
  const float* ln4_g = F(p++); const float* ln4_b = F(p++);
  const float* c5_w = F(p++); const float* c5_b = F(p++);
#undef F

  // ---- workspace carving (f32 region then f16 region), peak ~630 MB ----
  float* ws = (float*)d_ws;
  float* ssb  = ws;                 // 256
  float* inp  = ssb + 256;          // 32768
  float* hsb  = inp + 32768;        // 393216
  float* hfb  = hsb + 393216;
  float* hbb  = hfb + 393216;
  float* dten = hbb + 393216;       // 786432
  float* lnst = dten + 786432;      // 16
  const long BIG = 50331648;        // 8*64*64*1536
  float* big0 = lnst + 16;
  float* big1 = big0 + BIG;
  float* big2 = big1 + BIG;
  _Float16* hbase = (_Float16*)(big2 + BIG);
  long hoff = 0;
  auto halloc = [&](long n) { _Float16* q = hbase + hoff; hoff += (n + 7) & ~7L; return q; };

  _Float16* r1_wih16 = halloc(1024 * 256); _Float16* r1_whh16 = halloc(1024 * 256);
  _Float16* r2f_wih16 = halloc(1024 * 256); _Float16* r2f_whh16 = halloc(1024 * 256);
  _Float16* r2b_wih16 = halloc(1024 * 256); _Float16* r2b_whh16 = halloc(1024 * 256);
  _Float16* c0_p = halloc(256L * 3072); _Float16* c1_p = halloc(256L * 3072); _Float16* c2_p = halloc(256L * 3072);
  _Float16* rw1p[6]; _Float16* rw2p[6];
  const int resC[3] = {256, 128, 64};
  for (int s = 0; s < 3; ++s) {
    long sz = (long)resC[s] * resC[s] * 9;
    rw1p[2 * s] = halloc(sz); rw2p[2 * s] = halloc(sz);
    rw1p[2 * s + 1] = halloc(sz); rw2p[2 * s + 1] = halloc(sz);
  }
  _Float16* c3_p = halloc(128L * 2304);
  _Float16* c4_p = halloc(64L * 3200);
  _Float16* c5_p = halloc(16L * 3136);
  float* outp = (float*)d_out;

  // ---- weight preprocessing (f32 -> f16, flip baked in for convT) ----
  {
    const int n = 1024 * 256, blk = (n + 255) / 256;
    cvt_f16_kernel<<<blk, 256, 0, stream>>>(r1_wih, r1_wih16, n);
    cvt_f16_kernel<<<blk, 256, 0, stream>>>(r1_whh, r1_whh16, n);
    cvt_f16_kernel<<<blk, 256, 0, stream>>>(r2f_wih, r2f_wih16, n);
    cvt_f16_kernel<<<blk, 256, 0, stream>>>(r2f_whh, r2f_whh16, n);
    cvt_f16_kernel<<<blk, 256, 0, stream>>>(r2b_wih, r2b_wih16, n);
    cvt_f16_kernel<<<blk, 256, 0, stream>>>(r2b_whh, r2b_whh16, n);
  }
  launch_pack(stream, c0_w, c0_p, 256, 256, 4, 3, 1);
  launch_pack(stream, c1_w, c1_p, 256, 256, 4, 3, 1);
  launch_pack(stream, c2_w, c2_p, 256, 256, 4, 3, 1);
  for (int s = 0; s < 3; ++s)
    for (int b = 0; b < 2; ++b) {
      launch_pack(stream, res_w1[2 * s + b], rw1p[2 * s + b], resC[s], resC[s], 3, 3, 0);
      launch_pack(stream, res_w2[2 * s + b], rw2p[2 * s + b], resC[s], resC[s], 3, 3, 0);
    }
  launch_pack(stream, c3_w, c3_p, 128, 256, 3, 3, 0);
  launch_pack(stream, c4_w, c4_p, 64, 128, 5, 5, 0);
  launch_pack(stream, c5_w, c5_p, 1, 64, 7, 7, 0);

  // ---- front-end ----
  mlp_kernel<<<1, 256, 0, stream>>>(style, sp_w1, sp_b1, sp_w2, sp_b2, ssb);
  embed_kernel<<<(8 * 256 * 16 + 255) / 256, 256, 0, stream>>>(content, ssb, emb_w, emb_b, inp);

  // ---- sequential RNNs (WMMA gate GEMMs) ----
  spacing_rnn_kernel<<<1, 512, 0, stream>>>(inp, r1_wih16, r1_whh16, r1_bih, r1_bhh,
                                            shift_w, shift_b, hsb);
  bilstm_kernel<<<2, 512, 0, stream>>>(hsb, r2f_wih16, r2f_whh16, r2f_bih, r2f_bhh,
                                       r2b_wih16, r2b_whh16, r2b_bih, r2b_bhh, hfb, hbb);
  build_d_kernel<<<(8 * 256 * 384 + 255) / 256, 256, 0, stream>>>(hfb, hbb, dten);

  // ---- convT chain (flip baked into packed weights) ----
  launch_conv(stream, dten, c0_p, c0_b, nullptr, big0, 8, 256, 1, 384, 256, 4, 384,
              4, 3, 3, 1, 1, 1, 0, 1);
  launch_conv(stream, big0, c1_p, c1_b, nullptr, big1, 8, 256, 4, 384, 256, 8, 384,
              4, 3, 2, 1, 2, 1, 0, 1);
  launch_conv(stream, big1, c2_p, c2_b, nullptr, big0, 8, 256, 8, 384, 256, 16, 384,
              4, 3, 2, 1, 2, 1, 0, 1);

  // ---- res1 x2 @ 256ch, 16x384 ----  x=big0
  launch_conv(stream, big0, rw1p[0], res_b1[0], nullptr, big1, 8, 256, 16, 384, 256, 16, 384,
              3, 3, 1, 1, 1, 1, 0, 1);
  launch_conv(stream, big1, rw2p[0], res_b2[0], big0, big2, 8, 256, 16, 384, 256, 16, 384,
              3, 3, 1, 1, 1, 1, 0, 0);
  launch_conv(stream, big2, rw1p[1], res_b1[1], nullptr, big1, 8, 256, 16, 384, 256, 16, 384,
              3, 3, 1, 1, 1, 1, 0, 1);
  launch_conv(stream, big1, rw2p[1], res_b2[1], big2, big0, 8, 256, 16, 384, 256, 16, 384,
              3, 3, 1, 1, 1, 1, 0, 0);

  // ---- up2 (fused) + c3 + LN + relu ----
  launch_conv(stream, big0, c3_p, c3_b, nullptr, big1, 8, 256, 16, 384, 128, 32, 768,
              3, 3, 1, 1, 1, 1, 1, 0);
  {
    int n = 128 * 32 * 768;
    ln_stats_kernel<<<8, 256, 0, stream>>>(big1, lnst, n);
    ln_apply_kernel<<<(int)(((long)8 * n + 255) / 256), 256, 0, stream>>>(big1, lnst, ln3_g, ln3_b, 32 * 768, n);
  }

  // ---- res2 x2 @ 128ch, 32x768 ----  x=big1
  launch_conv(stream, big1, rw1p[2], res_b1[2], nullptr, big2, 8, 128, 32, 768, 128, 32, 768,
              3, 3, 1, 1, 1, 1, 0, 1);
  launch_conv(stream, big2, rw2p[2], res_b2[2], big1, big0, 8, 128, 32, 768, 128, 32, 768,
              3, 3, 1, 1, 1, 1, 0, 0);
  launch_conv(stream, big0, rw1p[3], res_b1[3], nullptr, big2, 8, 128, 32, 768, 128, 32, 768,
              3, 3, 1, 1, 1, 1, 0, 1);
  launch_conv(stream, big2, rw2p[3], res_b2[3], big0, big1, 8, 128, 32, 768, 128, 32, 768,
              3, 3, 1, 1, 1, 1, 0, 0);

  // ---- up2 (fused) + c4 + LN + relu ----
  launch_conv(stream, big1, c4_p, c4_b, nullptr, big0, 8, 128, 32, 768, 64, 64, 1536,
              5, 5, 2, 2, 1, 1, 1, 0);
  {
    int n = 64 * 64 * 1536;
    ln_stats_kernel<<<8, 256, 0, stream>>>(big0, lnst, n);
    ln_apply_kernel<<<(int)(((long)8 * n + 255) / 256), 256, 0, stream>>>(big0, lnst, ln4_g, ln4_b, 64 * 1536, n);
  }

  // ---- res3 x2 @ 64ch, 64x1536 ----  x=big0
  launch_conv(stream, big0, rw1p[4], res_b1[4], nullptr, big1, 8, 64, 64, 1536, 64, 64, 1536,
              3, 3, 1, 1, 1, 1, 0, 1);
  launch_conv(stream, big1, rw2p[4], res_b2[4], big0, big2, 8, 64, 64, 1536, 64, 64, 1536,
              3, 3, 1, 1, 1, 1, 0, 0);
  launch_conv(stream, big2, rw1p[5], res_b1[5], nullptr, big1, 8, 64, 64, 1536, 64, 64, 1536,
              3, 3, 1, 1, 1, 1, 0, 1);
  launch_conv(stream, big1, rw2p[5], res_b2[5], big2, big0, 8, 64, 64, 1536, 64, 64, 1536,
              3, 3, 1, 1, 1, 1, 0, 0);

  // ---- c5 7x7 pad3 -> d_out, tanh ----
  launch_conv(stream, big0, c5_p, c5_b, nullptr, outp, 8, 64, 64, 1536, 1, 64, 1536,
              7, 7, 3, 3, 1, 1, 0, 2);
}
